// Attention_84542136254713
// MI455X (gfx1250) — matrix operand order repaired and checked by reading, not compiled, not run on hardware
//
#include <hip/hip_runtime.h>

#define Bsz 8
#define Ssz 2048
#define Dsz 1024
#define NEGC 100000000000.0f

typedef __bf16 bf16;
typedef __attribute__((ext_vector_type(8)))  __bf16 v8bf;
typedef __attribute__((ext_vector_type(16))) __bf16 v16bf;
typedef __attribute__((ext_vector_type(8)))  float  v8f;

// ---------------- fragment loaders (wave32 WMMA bf16 layouts) ----------------
// A 16x32 bf16: lane L -> row M = L&15; elems 0..7 = K(kb..kb+7), 8..15 = K(kb+16..kb+23), kb = k0 + 8*(L>>4)
__device__ __forceinline__ v16bf load_a_frag(const bf16* row, int k0, int half) {
    int kb = k0 + 8 * half;
    v8bf lo = *(const v8bf*)(row + kb);
    v8bf hi = *(const v8bf*)(row + kb + 16);
    v16bf r;
#pragma unroll
    for (int i = 0; i < 8; ++i) { r[i] = lo[i]; r[8 + i] = hi[i]; }
    return r;
}
// B 32x16 bf16: lane L -> col N = L&15; elems 0..7 = K(kb..kb+7), 8..15 = K(kb+16..kb+23), kb = k0 + 8*(L>>4)
__device__ __forceinline__ v16bf load_b_frag(const bf16* row, int k0, int half) {
    int kb = k0 + 8 * half;
    v8bf lo = *(const v8bf*)(row + kb);
    v8bf hi = *(const v8bf*)(row + kb + 16);
    v16bf r;
#pragma unroll
    for (int i = 0; i < 8; ++i) { r[i] = lo[i]; r[8 + i] = hi[i]; }
    return r;
}

__device__ __forceinline__ v8f wmma_bf16(v16bf a, v16bf b, v8f c) {
    return __builtin_amdgcn_wmma_f32_16x16x32_bf16(false, a, false, b, (short)0, c, false, false);
}

// One wave computes a 32(M) x 64(N) f32 tile. Single-buffered fragments; all 12
// b128 loads of a K-step are issued before the 8 wmmas so the compiler can
// clause them; cross-wave occupancy (~8 waves/SIMD) hides the latency.
// acc[rt*4 + t]: rt = M-subtile, t = N-subtile.
__device__ __forceinline__ void wave_gemm_32x64(const bf16* A, int lda,
                                                const bf16* Bm, int ldb,
                                                int m0, int n0, int K, v8f* acc) {
    const int lane = threadIdx.x & 31;
    const int half = lane >> 4;
    const int lr   = lane & 15;
    const bf16* arow0 = A  + (size_t)(m0 + lr) * lda;
    const bf16* arow1 = A  + (size_t)(m0 + 16 + lr) * lda;
    const bf16* brow  = Bm + (size_t)(n0 + lr) * ldb;

    for (int k0 = 0; k0 < K; k0 += 32) {
        v16bf a0 = load_a_frag(arow0, k0, half);
        v16bf a1 = load_a_frag(arow1, k0, half);
        v16bf b0 = load_b_frag(brow,                      k0, half);
        v16bf b1 = load_b_frag(brow + (size_t)16 * ldb,   k0, half);
        v16bf b2 = load_b_frag(brow + (size_t)32 * ldb,   k0, half);
        v16bf b3 = load_b_frag(brow + (size_t)48 * ldb,   k0, half);
        acc[0] = wmma_bf16(a0, b0, acc[0]);
        acc[4] = wmma_bf16(a1, b0, acc[4]);
        acc[1] = wmma_bf16(a0, b1, acc[1]);
        acc[5] = wmma_bf16(a1, b1, acc[5]);
        acc[2] = wmma_bf16(a0, b2, acc[2]);
        acc[6] = wmma_bf16(a1, b2, acc[6]);
        acc[3] = wmma_bf16(a0, b3, acc[3]);
        acc[7] = wmma_bf16(a1, b3, acc[7]);
    }
}

// ---------------- pass 1: f32 -> bf16 conversions ----------------
__global__ void __launch_bounds__(256) cvt_x_kernel(const float* __restrict__ x,
                                                    bf16* __restrict__ xbf,
                                                    bf16* __restrict__ xT) {
    size_t idx = (size_t)blockIdx.x * blockDim.x + threadIdx.x; // B*S*D threads
    float v = x[idx];
    bf16 h = (bf16)v;
    xbf[idx] = h;
    size_t d  = idx % Dsz;
    size_t bs = idx / Dsz;
    size_t s  = bs % Ssz;
    size_t b  = bs / Ssz;
    xT[(b * Dsz + d) * Ssz + s] = h; // xT[b][d][s]
}

__global__ void __launch_bounds__(256) cvt_w_kernel(const float* __restrict__ w,
                                                    bf16* __restrict__ wT) {
    size_t idx = (size_t)blockIdx.x * blockDim.x + threadIdx.x; // D*D threads
    size_t k = idx / Dsz, n = idx % Dsz;
    wT[n * Dsz + k] = (bf16)w[idx]; // wT[n][k] = W[k][n]
}

// ---------------- pass 2: xw = x @ W  (M=B*S, N=D, K=D) ----------------
__global__ void __launch_bounds__(256) gemm_xw_kernel(const bf16* __restrict__ xbf,
                                                      const bf16* __restrict__ wT,
                                                      bf16* __restrict__ xw) {
    const int wave = threadIdx.x >> 5;
    const int m0 = blockIdx.x * 32;
    const int n0 = (blockIdx.y * 8 + wave) * 64;
    v8f acc[8] = {};
    wave_gemm_32x64(xbf, Dsz, wT, Dsz, m0, n0, Dsz, acc);
    const int lane = threadIdx.x & 31, half = lane >> 4, lr = lane & 15;
#pragma unroll
    for (int rt = 0; rt < 2; ++rt)
#pragma unroll
        for (int t = 0; t < 4; ++t) {
            int n = n0 + t * 16 + lr;
#pragma unroll
            for (int r = 0; r < 8; ++r) {
                int m = m0 + rt * 16 + r + 8 * half;
                xw[(size_t)m * Dsz + n] = (bf16)acc[rt * 4 + t][r];
            }
        }
}

// ---------------- pass 3: alpha = xw @ x^T + bias + (mask-1)*NEG ----------------
__global__ void __launch_bounds__(256) gemm_scores_kernel(const bf16* __restrict__ xw,
                                                          const bf16* __restrict__ xbf,
                                                          const float* __restrict__ mask,
                                                          const float* __restrict__ bias,
                                                          bf16* __restrict__ alpha) {
    const int b = blockIdx.z;
    const int wave = threadIdx.x >> 5;
    const int m0 = blockIdx.x * 32;
    const int n0 = (blockIdx.y * 8 + wave) * 64;
    const bf16* A  = xw  + (size_t)b * Ssz * Dsz;
    const bf16* Bm = xbf + (size_t)b * Ssz * Dsz; // B(k,j) = x[b][j][k] -> row j contiguous in k
    v8f acc[8] = {};
    wave_gemm_32x64(A, Dsz, Bm, Dsz, m0, n0, Dsz, acc);
    const int lane = threadIdx.x & 31, half = lane >> 4, lr = lane & 15;
    const float bs = bias[0];
    bf16* outp = alpha + (size_t)b * Ssz * Ssz;
#pragma unroll
    for (int t = 0; t < 4; ++t) {
        int n = n0 + t * 16 + lr;
        float mt = (mask[(size_t)b * Ssz + n] - 1.0f) * NEGC + bs;
#pragma unroll
        for (int rt = 0; rt < 2; ++rt)
#pragma unroll
            for (int r = 0; r < 8; ++r) {
                int m = m0 + rt * 16 + r + 8 * half;
                outp[(size_t)m * Ssz + n] = (bf16)(acc[rt * 4 + t][r] + mt);
            }
    }
}

// ---------------- pass 4: row softmax over S (in-place on bf16 alpha) ----------------
__global__ void __launch_bounds__(256) softmax_kernel(bf16* __restrict__ alpha) {
    const size_t row = blockIdx.x; // B*S rows
    bf16* p = alpha + row * (size_t)Ssz;
    __shared__ float red[256];
    const int tid = threadIdx.x;
    float vals[8];
    float lmax = -3.0e38f;
#pragma unroll
    for (int i = 0; i < 8; ++i) {
        vals[i] = (float)p[tid + i * 256];
        lmax = fmaxf(lmax, vals[i]);
    }
    red[tid] = lmax;
    __syncthreads();
    for (int off = 128; off > 0; off >>= 1) {
        if (tid < off) red[tid] = fmaxf(red[tid], red[tid + off]);
        __syncthreads();
    }
    float m = red[0];
    __syncthreads();
    float lsum = 0.0f;
#pragma unroll
    for (int i = 0; i < 8; ++i) {
        float e = __expf(vals[i] - m);
        vals[i] = e;
        lsum += e;
    }
    red[tid] = lsum;
    __syncthreads();
    for (int off = 128; off > 0; off >>= 1) {
        if (tid < off) red[tid] += red[tid + off];
        __syncthreads();
    }
    float inv = 1.0f / red[0];
#pragma unroll
    for (int i = 0; i < 8; ++i)
        p[tid + i * 256] = (bf16)(vals[i] * inv);
}

// ---------------- pass 5: context = alpha @ x  (M=S, N=D, K=S) ----------------
__global__ void __launch_bounds__(256) gemm_ctx_kernel(const bf16* __restrict__ alpha,
                                                       const bf16* __restrict__ xT,
                                                       float* __restrict__ out) {
    const int b = blockIdx.z;
    const int wave = threadIdx.x >> 5;
    const int m0 = blockIdx.x * 32;
    const int n0 = (blockIdx.y * 8 + wave) * 64;
    const bf16* A  = alpha + (size_t)b * Ssz * Ssz;          // lda = S
    const bf16* Bm = xT    + (size_t)b * Dsz * Ssz;          // B(k,d)=x[b][k][d]=xT[b][d][k], ldb = S
    v8f acc[8] = {};
    wave_gemm_32x64(A, Ssz, Bm, Ssz, m0, n0, Ssz, acc);
    const int lane = threadIdx.x & 31, half = lane >> 4, lr = lane & 15;
    float* outp = out + (size_t)b * Ssz * Dsz;
#pragma unroll
    for (int rt = 0; rt < 2; ++rt)
#pragma unroll
        for (int t = 0; t < 4; ++t) {
            int n = n0 + t * 16 + lr;
#pragma unroll
            for (int r = 0; r < 8; ++r) {
                int m = m0 + rt * 16 + r + 8 * half;
                outp[(size_t)m * Dsz + n] = acc[rt * 4 + t][r];
            }
        }
}

extern "C" void kernel_launch(void* const* d_in, const int* in_sizes, int n_in,
                              void* d_out, int out_size, void* d_ws, size_t ws_size,
                              hipStream_t stream) {
    const float* x    = (const float*)d_in[0];
    const float* mask = (const float*)d_in[1];
    const float* w    = (const float*)d_in[2];
    const float* bias = (const float*)d_in[3];
    float* out = (float*)d_out;

    char* ws = (char*)d_ws;
    // workspace layout (bytes)
    bf16* xbf   = (bf16*)(ws);                                 // B*S*D*2   = 33,554,432
    bf16* xT    = (bf16*)(ws + (size_t)33554432);              // B*D*S*2   = 33,554,432
    bf16* wT    = (bf16*)(ws + (size_t)67108864);              // D*D*2     =  2,097,152
    bf16* xw    = (bf16*)(ws + (size_t)69206016);              // B*S*D*2   = 33,554,432
    bf16* alpha = (bf16*)(ws + (size_t)102760448);             // B*S*S*2   = 67,108,864
    // total 169,869,312 bytes

    // pass 1: conversions
    cvt_x_kernel<<<(Bsz * Ssz * Dsz) / 256, 256, 0, stream>>>(x, xbf, xT);
    cvt_w_kernel<<<(Dsz * Dsz) / 256, 256, 0, stream>>>(w, wT);

    // pass 2: xw = x @ W   (M = B*S = 16384, N = 1024)
    gemm_xw_kernel<<<dim3(512, 2, 1), 256, 0, stream>>>(xbf, wT, xw);

    // pass 3: alpha = xw @ x^T + bias + mask  (per batch, M=N=2048)
    gemm_scores_kernel<<<dim3(64, 4, Bsz), 256, 0, stream>>>(xw, xbf, mask, bias, alpha);

    // pass 4: softmax over rows
    softmax_kernel<<<dim3(Bsz * Ssz, 1, 1), 256, 0, stream>>>(alpha);

    // pass 5: context = alpha @ x  (per batch, M=2048, N=1024)
    gemm_ctx_kernel<<<dim3(64, 2, Bsz), 256, 0, stream>>>(alpha, xT, out);
}
